// QuantizedLinear_75385265979770
// MI455X (gfx1250) — 8-bit matrix operand order repaired and checked by reading, compiled, not run on hardware
//
#include <hip/hip_runtime.h>


typedef int v8i __attribute__((ext_vector_type(8)));

#define K_DIM 256
#define O_DIM 256
#define TOKENS_PER_BLOCK 128
#define BLOCK_THREADS 256

__device__ __forceinline__ int clamp255(int q) {
    q = q < 0 ? 0 : q;
    return q > 255 ? 255 : q;
}

__device__ __forceinline__ unsigned int quant4(float4 f) {
    int q0 = clamp255(__float2int_rn(__builtin_fmaf(f.x, 127.5f, 128.0f))) - 128;
    int q1 = clamp255(__float2int_rn(__builtin_fmaf(f.y, 127.5f, 128.0f))) - 128;
    int q2 = clamp255(__float2int_rn(__builtin_fmaf(f.z, 127.5f, 128.0f))) - 128;
    int q3 = clamp255(__float2int_rn(__builtin_fmaf(f.w, 127.5f, 128.0f))) - 128;
    return (unsigned int)(q0 & 0xFF) | ((unsigned int)(q1 & 0xFF) << 8) |
           ((unsigned int)(q2 & 0xFF) << 16) | ((unsigned int)(q3 & 0xFF) << 24);
}

__global__ __launch_bounds__(BLOCK_THREADS)
void qlinear_i8_wmma(const float* __restrict__ x,
                     const float* __restrict__ w,
                     const float* __restrict__ b,
                     float* __restrict__ out,
                     int nTokens) {
    __shared__ unsigned int wLDS[16384];
    __shared__ unsigned int xLDS[8192];

    const int tid  = threadIdx.x;
    const int lane = tid & 31;
    const int wave = tid >> 5;
    const int hi   = (lane >= 16) ? 1 : 0;
    const int tokenBase = blockIdx.x * TOKENS_PER_BLOCK;
    const int rowBase   = tokenBase + wave * 16;

    if (tokenBase + TOKENS_PER_BLOCK < nTokens) {
        const float* nx = x + (size_t)(tokenBase + TOKENS_PER_BLOCK) * K_DIM;
        __builtin_prefetch(nx + tid * 16, 0, 0);
    }

#pragma unroll
    for (int it = 0; it < 64; ++it) {
        int wi = tid + it * BLOCK_THREADS;
        int l  = wi & 31;
        int j  = (wi >> 5) & 7;
        int c  = (wi >> 8) & 3;
        int t  = (wi >> 10) & 15;
        int o  = t * 16 + (l & 15);
        int k0 = c * 64 + (j & 3) * 4 + ((l >= 16) ? 16 : 0) + ((j >= 4) ? 32 : 0);
        float4 wf = *(const float4*)(w + o * K_DIM + k0);
        unsigned int p = (unsigned int)(__float2int_rn(wf.x) & 0xFF)
                       | ((unsigned int)(__float2int_rn(wf.y) & 0xFF) << 8)
                       | ((unsigned int)(__float2int_rn(wf.z) & 0xFF) << 16)
                       | ((unsigned int)(__float2int_rn(wf.w) & 0xFF) << 24);
        wLDS[wi] = p;
    }

    unsigned int* xw = xLDS + wave * 1024;
#pragma unroll
    for (int it = 0; it < 32; ++it) {
        int wi = lane + it * 32;
        int m  = wi >> 6;
        int kw = wi & 63;
        float4 xf = *(const float4*)(x + (size_t)(rowBase + m) * K_DIM + kw * 4);
        xw[wi] = quant4(xf);
    }

    __syncthreads();

    v8i a[4];
#pragma unroll
    for (int c = 0; c < 4; ++c) {
#pragma unroll
        for (int j = 0; j < 8; ++j) {
            int k0 = c * 64 + (j & 3) * 4 + (hi ? 16 : 0)
                   + ((j >= 4) ? 32 : 0);
            a[c][j] = (int)xw[(lane & 15) * 64 + (k0 >> 2)];
        }
    }

    const float scale = (2.0f / 255.0f) * 0.01f;

    for (int t = 0; t < 16; ++t) {
        v8i acc = {0, 0, 0, 0, 0, 0, 0, 0};
#pragma unroll
        for (int c = 0; c < 4; ++c) {
            v8i bm;
#pragma unroll
            for (int j = 0; j < 8; ++j)
                bm[j] = (int)wLDS[(((t * 4 + c) * 8) + j) * 32 + lane];
            acc = __builtin_amdgcn_wmma_i32_16x16x64_iu8(
                true, a[c], true, bm, acc, false, false);
        }

        int   o    = t * 16 + (lane & 15);
        float bias = b[o] * 0.001f;
        int   mRow = rowBase + (hi ? 8 : 0);
#pragma unroll
        for (int r = 0; r < 8; ++r) {
            out[(size_t)(mRow + r) * O_DIM + o] =
                (float)acc[r] * scale + bias;
        }
    }
}

extern "C" void kernel_launch(void* const* d_in, const int* in_sizes, int n_in,
                              void* d_out, int out_size, void* d_ws, size_t ws_size,
                              hipStream_t stream) {
    const float* x = (const float*)d_in[0];
    const float* w = (const float*)d_in[1];
    const float* b = (const float*)d_in[2];
    float* out = (float*)d_out;

    const int nTokens = in_sizes[0] / K_DIM;
    const int blocks  = nTokens / TOKENS_PER_BLOCK;

    qlinear_i8_wmma<<<blocks, BLOCK_THREADS, 0, stream>>>(x, w, b, out, nTokens);
}
